// GeometricLinearizedAttention_89232240541712
// MI455X (gfx1250) — matrix operand order repaired and checked by reading, not compiled, not run on hardware
//
#include <hip/hip_runtime.h>
#include <hip/hip_bf16.h>
#include <math.h>

typedef __attribute__((ext_vector_type(16))) __bf16 v16bf;
typedef __attribute__((ext_vector_type(8)))  __bf16 v8bf;
typedef __attribute__((ext_vector_type(2)))  __bf16 v2bf;
typedef __attribute__((ext_vector_type(8)))  float  v8f;

#define GLA_N 4096
#define GLA_D 64
#define GLA_H 16
#define CHUNK 128
#define NWAVE 16
#define TPB   512
#define RPW   (CHUNK / NWAVE)   // rows per wave in elementwise stages = 8

#define EPSV      1e-5f
#define PROJ_EPSF 0.004f
#define MIN_NRM   1e-15f

// ---- LDS layout (byte offsets, all 16B aligned) ----
#define V2T_STR 136   // bf16 halves per row (128 + 8 pad)
#define XT_STR  136
#define V1_STR  72    // 64 + 8 pad
#define CTX_STR 72
#define XO_STR  68    // 64 + 4 pad (f32)

#define OFF_FAC   0        // float [4096]            16384 B
#define OFF_V2T   16384    // bf16  [64][136]         17408 B
#define OFF_XT    33792    // bf16  [64][136]         17408 B
#define OFF_V1    51200    // bf16  [128][72]         18432 B
#define OFF_CTX   69632    // bf16  [64][72]           9216 B
#define OFF_XO    78848    // float [128][68]         34816 B
#define OFF_V2SUM 113664   // float [64]                256 B
#define OFF_WSUM  113920   // float [16][64]           4096 B
#define OFF_DINV  118016   // float [128]               512 B
#define SMEM_BYTES 118528

// ---------- fast scalar helpers ----------
__device__ __forceinline__ float frcp(float x)  { return __builtin_amdgcn_rcpf(x); }
__device__ __forceinline__ float fsqrt(float x) { return __builtin_amdgcn_sqrtf(x); }

// elu(x) + 1  ==  x+1 (x>0)  else  exp(x)
__device__ __forceinline__ float elu_p1(float x) {
    return (x > 0.0f) ? (x + 1.0f) : __expf(x);
}
__device__ __forceinline__ float atanh_fast(float y) {        // y in [0, 1)
    return 0.5f * __logf((1.0f + y) * frcp(1.0f - y));
}
__device__ __forceinline__ float tanh_fast(float t) {         // t >= 0
    return 1.0f - 2.0f * frcp(__expf(2.0f * t) + 1.0f);
}

// ---------- VALU-only 32-lane xor-butterfly sum (no LDS traffic) ----------
template <int CTRL>
__device__ __forceinline__ float dpp_xor_add(float v) {
    int p = __builtin_amdgcn_update_dpp(0, __float_as_int(v), CTRL, 0xF, 0xF, true);
    return v + __int_as_float(p);
}
__device__ __forceinline__ float wave_sum32(float v) {
    v = dpp_xor_add<0xB1>(v);    // quad_perm [1,0,3,2]  -> xor 1
    v = dpp_xor_add<0x4E>(v);    // quad_perm [2,3,0,1]  -> xor 2
    v = dpp_xor_add<0x164>(v);   // row_xmask: 4         -> xor 4
    v = dpp_xor_add<0x168>(v);   // row_xmask: 8         -> xor 8
    int s = __float_as_int(v);   // xor 16: swap 16-lane halves, identity select
    int p = __builtin_amdgcn_permlanex16(s, s, 0x76543210, 0xFEDCBA98, true, false);
    return v + __int_as_float(p);
}

__device__ __forceinline__ v16bf cat8(v8bf lo, v8bf hi) {
    return __builtin_shufflevector(lo, hi, 0, 1, 2, 3, 4, 5, 6, 7,
                                           8, 9, 10, 11, 12, 13, 14, 15);
}

__global__ __launch_bounds__(TPB)
void gla_wmma_kernel(const float* __restrict__ Q, const float* __restrict__ K,
                     const float* __restrict__ V, const float* __restrict__ mask,
                     const float* __restrict__ c, int csize,
                     float* __restrict__ out)
{
    extern __shared__ char smem[];
    float*  s_fac   = (float*)(smem + OFF_FAC);
    __bf16* s_v2t   = (__bf16*)(smem + OFF_V2T);
    __bf16* s_xt    = (__bf16*)(smem + OFF_XT);
    __bf16* s_v1    = (__bf16*)(smem + OFF_V1);
    __bf16* s_ctx   = (__bf16*)(smem + OFF_CTX);
    float*  s_xo    = (float*)(smem + OFF_XO);
    float*  s_v2sum = (float*)(smem + OFF_V2SUM);
    float*  s_wsum  = (float*)(smem + OFF_WSUM);
    float*  s_dinv  = (float*)(smem + OFF_DINV);

    const int bh = blockIdx.x;
    const int h  = bh % GLA_H;
    const float k = (csize == 1) ? c[0] : c[h];

    const int wave = threadIdx.x >> 5;
    const int lane = threadIdx.x & 31;

    const float* Qb = Q + (size_t)bh * GLA_N * GLA_D;
    const float* Kb = K + (size_t)bh * GLA_N * GLA_D;
    const float* Vb = V + (size_t)bh * GLA_N * GLA_D;
    float*       Ob = out + (size_t)bh * GLA_N * GLA_D;

    // WMMA lane geometry (ISA 05_wmma 16-bit layouts, wave32)
    const int lm   = lane & 15;   // tile-row (A) / tile-col (B,C)
    const int lgrp = lane >> 4;   // lane half
    const int kgA  = lgrp * 8;    // A frag: K offsets {0..7}+16 vs {8..15}+16
    const int kgB  = lgrp * 8;    // B frag: K offsets {0..7}+16 vs {8..15}+16

    float v2s0 = 0.0f, v2s1 = 0.0f;
    v8f ctx_acc = {};                       // this wave's 16x16 context tile
    const int mt1 = wave >> 2, nt1 = wave & 3;

    // ======================= Phase 1: context = v2^T * x =======================
    for (int chunk = 0; chunk < GLA_N; chunk += CHUNK) {
        const bool pf = (chunk + CHUNK) < GLA_N;
#pragma unroll
        for (int r = 0; r < RPW; ++r) {
            const int rl = wave * RPW + r;
            const int n  = chunk + rl;
            if (pf) {   // prefetch next chunk's cachelines (global_prefetch_b8)
                __builtin_prefetch(Vb + (size_t)(n + CHUNK) * GLA_D + 2 * lane, 0, 1);
                __builtin_prefetch(Kb + (size_t)(n + CHUNK) * GLA_D + 2 * lane, 0, 1);
            }
            const float2 vv = *(const float2*)(Vb + (size_t)n * GLA_D + 2 * lane);
            const float ss = wave_sum32(vv.x * vv.x + vv.y * vv.y);
            const float fac   = frcp(fmaxf(1.0f + k * ss, MIN_NRM));
            const float gamma = 2.0f * fac;
            const float gm1   = gamma - 1.0f;
            const float sgn   = (gm1 >= 0.0f) ? 1.0f : -1.0f;
            const float denom = sgn * fmaxf(fabsf(gm1), 1e-10f);
            const float mv    = mask[n];
            if (lane == 0) s_fac[n] = fac;
            const float xs = gamma * frcp(denom) * mv;
            const float2 kv = *(const float2*)(Kb + (size_t)n * GLA_D + 2 * lane);
            const float v2a = denom * elu_p1(kv.x * fac) * mv;
            const float v2b = denom * elu_p1(kv.y * fac) * mv;
            v2s0 += v2a; v2s1 += v2b;
            // transposed stage: [d][n_local] so WMMA frags are contiguous b128s
            s_v2t[(2 * lane    ) * V2T_STR + rl] = (__bf16)v2a;
            s_v2t[(2 * lane + 1) * V2T_STR + rl] = (__bf16)v2b;
            s_xt [(2 * lane    ) * XT_STR  + rl] = (__bf16)(xs * vv.x);
            s_xt [(2 * lane + 1) * XT_STR  + rl] = (__bf16)(xs * vv.y);
        }
        __syncthreads();
        {
            const __bf16* arow = s_v2t + (mt1 * 16 + lm) * V2T_STR;  // A: M=d, K=n
            const __bf16* brow = s_xt  + (nt1 * 16 + lm) * XT_STR;   // B: K=n, N=e
#pragma unroll
            for (int kk = 0; kk < CHUNK; kk += 32) {
                v8bf alo = *(const v8bf*)(arow + kk + kgA);
                v8bf ahi = *(const v8bf*)(arow + kk + kgA + 16);
                v8bf blo = *(const v8bf*)(brow + kk + kgB);
                v8bf bhi = *(const v8bf*)(brow + kk + kgB + 16);
                ctx_acc = __builtin_amdgcn_wmma_f32_16x16x32_bf16(
                    false, cat8(alo, ahi), false, cat8(blo, bhi),
                    (short)0, ctx_acc, false, false);
            }
        }
        __syncthreads();
    }

    // ---- v2_sum cross-wave reduction; context -> bf16 transposed ----
    s_wsum[wave * GLA_D + 2 * lane]     = v2s0;
    s_wsum[wave * GLA_D + 2 * lane + 1] = v2s1;
    __syncthreads();
    if (threadIdx.x < GLA_D) {
        float t = 0.0f;
#pragma unroll
        for (int w = 0; w < NWAVE; ++w) t += s_wsum[w * GLA_D + threadIdx.x];
        s_v2sum[threadIdx.x] = t;
    }
    {
        const int e  = nt1 * 16 + lm;        // C layout: N = lane&15
        const int d0 = mt1 * 16 + lgrp * 8;  // C layout: M = r + 8*(lane>>4)
        v8bf cv;
#pragma unroll
        for (int r = 0; r < 8; ++r) cv[r] = (__bf16)ctx_acc[r];
        *(v8bf*)(s_ctx + e * CTX_STR + d0) = cv;   // ctx^T[e][d]
    }
    __syncthreads();

    // ======================= Phase 2: X = (v1 @ ctx) * Dinv, project, mobius ====
    const float sk     = fsqrt(fabsf(k) + MIN_NRM);
    const float inv_sk = frcp(sk);
    const float maxn   = (k < 0.0f) ? (1.0f - PROJ_EPSF) * inv_sk : 1e15f;

    for (int chunk = 0; chunk < GLA_N; chunk += CHUNK) {
        const bool pf = (chunk + CHUNK) < GLA_N;
#pragma unroll
        for (int r = 0; r < RPW; ++r) {
            const int rl = wave * RPW + r;
            const int n  = chunk + rl;
            if (pf)
                __builtin_prefetch(Qb + (size_t)(n + CHUNK) * GLA_D + 2 * lane, 0, 1);
            const float fac = s_fac[n];
            const float2 qv = *(const float2*)(Qb + (size_t)n * GLA_D + 2 * lane);
            const float v1a = elu_p1(qv.x * fac);
            const float v1b = elu_p1(qv.y * fac);
            v2bf pr; pr[0] = (__bf16)v1a; pr[1] = (__bf16)v1b;
            *(v2bf*)(s_v1 + rl * V1_STR + 2 * lane) = pr;   // row-major: A-ready
            const float dn = wave_sum32(v1a * s_v2sum[2 * lane] +
                                        v1b * s_v2sum[2 * lane + 1]);
            if (lane == 0) s_dinv[rl] = frcp((dn == 0.0f) ? EPSV : dn);
        }
        __syncthreads();
#pragma unroll
        for (int tt = 0; tt < 2; ++tt) {
            const int t  = wave + tt * NWAVE;    // 32 tiles: 8 row x 4 col
            const int mt = t >> 2, nt = t & 3;
            v8f acc = {};
            const __bf16* arow = s_v1  + (mt * 16 + lm) * V1_STR;   // A: M=n, K=d
            const __bf16* brow = s_ctx + (nt * 16 + lm) * CTX_STR;  // B: K=d, N=e
#pragma unroll
            for (int kk = 0; kk < GLA_D; kk += 32) {
                v8bf alo = *(const v8bf*)(arow + kk + kgA);
                v8bf ahi = *(const v8bf*)(arow + kk + kgA + 16);
                v8bf blo = *(const v8bf*)(brow + kk + kgB);
                v8bf bhi = *(const v8bf*)(brow + kk + kgB + 16);
                acc = __builtin_amdgcn_wmma_f32_16x16x32_bf16(
                    false, cat8(alo, ahi), false, cat8(blo, bhi),
                    (short)0, acc, false, false);
            }
            const int row0 = mt * 16 + lgrp * 8;
            const int col  = nt * 16 + lm;
#pragma unroll
            for (int r = 0; r < 8; ++r)
                s_xo[(row0 + r) * XO_STR + col] = acc[r] * s_dinv[row0 + r];
        }
        __syncthreads();
#pragma unroll
        for (int r = 0; r < RPW; ++r) {
            const int rl = wave * RPW + r;
            const int n  = chunk + rl;
            const float xa = s_xo[rl * XO_STR + 2 * lane];
            const float xb = s_xo[rl * XO_STR + 2 * lane + 1];
            const float nrm = fmaxf(fsqrt(wave_sum32(xa * xa + xb * xb)), MIN_NRM);
            float sc = (nrm > maxn) ? (maxn * frcp(nrm)) : 1.0f;     // project #1
            const float xn = fmaxf(nrm * sc, MIN_NRM);
            // mobius_scalar_mul(0.5): sk factors folded — u = 0.5*artan_k-core
            const float u  = (k < 0.0f)
                               ? 0.5f * atanh_fast(fminf(sk * xn, 1.0f - 1e-7f))
                               : 0.5f * atanf(sk * xn);
            const float tv = ((k < 0.0f) ? tanh_fast(u) : tanf(u)) * inv_sk;
            sc *= tv * frcp(xn);
            const float n2 = fmaxf(fabsf(tv), MIN_NRM);
            if (n2 > maxn) sc *= maxn * frcp(n2);                    // project #2
            float2 ov; ov.x = xa * sc; ov.y = xb * sc;
            *(float2*)(Ob + (size_t)n * GLA_D + 2 * lane) = ov;
        }
        __syncthreads();
    }
}

extern "C" void kernel_launch(void* const* d_in, const int* in_sizes, int n_in,
                              void* d_out, int out_size, void* d_ws, size_t ws_size,
                              hipStream_t stream) {
    (void)n_in; (void)out_size; (void)d_ws; (void)ws_size;
    const float* Q    = (const float*)d_in[0];
    const float* K    = (const float*)d_in[1];
    const float* V    = (const float*)d_in[2];
    const float* mask = (const float*)d_in[3];
    const float* c    = (const float*)d_in[4];
    const int csize = in_sizes[4];
    const int BH    = in_sizes[0] / (GLA_N * GLA_D);   // B*H = 64

    (void)hipFuncSetAttribute((const void*)gla_wmma_kernel,
                              hipFuncAttributeMaxDynamicSharedMemorySize,
                              SMEM_BYTES);
    gla_wmma_kernel<<<dim3(BH), dim3(TPB), SMEM_BYTES, stream>>>(
        Q, K, V, mask, c, csize, (float*)d_out);
}
